// NNMatcher_6021544149480
// MI455X (gfx1250) — matrix operand order repaired and checked by reading, not compiled, not run on hardware
//
#include <hip/hip_runtime.h>

// Problem constants (match reference): B=4, N=8192, D=64
#define BB 4
#define NN 8192
#define DD 64

typedef __attribute__((ext_vector_type(16))) _Float16 v16h;
typedef __attribute__((ext_vector_type(8)))  _Float16 v8h;
typedef __attribute__((ext_vector_type(8)))  float    v8f;

// ---------------------------------------------------------------------------
// Kernel 1: L2-normalize descriptors -> f16 row-major in workspace.
// gid in [0, 2*B*N): first half handles desc_r rows (also copies xy_r to out),
// second half handles desc_n rows.
// ---------------------------------------------------------------------------
__global__ void __launch_bounds__(256)
nn_pack(const float* __restrict__ xy_r,
        const float* __restrict__ desc_r,
        const float* __restrict__ desc_n,
        _Float16* __restrict__ ha,
        _Float16* __restrict__ hb,
        float* __restrict__ out_xyr)
{
    int gid = blockIdx.x * blockDim.x + threadIdx.x;
    if (gid >= 2 * BB * NN) return;

    const int  sel = (gid >= BB * NN) ? 1 : 0;       // 0 = desc_r, 1 = desc_n
    const int  row = gid - sel * (BB * NN);          // flat row over [B*N)

    if (!sel) {
        // pass-through xy_r -> first output region
        const float2* xr = (const float2*)xy_r;
        ((float2*)out_xyr)[row] = xr[row];
    }

    const float* src = sel ? desc_n : desc_r;
    _Float16*    dst = sel ? hb : ha;

    const float4* sp = (const float4*)(src + (size_t)row * DD);
    float4 v[16];
    float s = 0.f;
#pragma unroll
    for (int i = 0; i < 16; ++i) {
        v[i] = sp[i];
        s += v[i].x * v[i].x + v[i].y * v[i].y + v[i].z * v[i].z + v[i].w * v[i].w;
    }
    const float inv = 1.0f / sqrtf(s);   // matches x / ||x||

    v8h* dp = (v8h*)(dst + (size_t)row * DD);
#pragma unroll
    for (int i = 0; i < 8; ++i) {
        float4 a = v[2 * i], b = v[2 * i + 1];
        v8h o = { (_Float16)(a.x * inv), (_Float16)(a.y * inv),
                  (_Float16)(a.z * inv), (_Float16)(a.w * inv),
                  (_Float16)(b.x * inv), (_Float16)(b.y * inv),
                  (_Float16)(b.z * inv), (_Float16)(b.w * inv) };
        dp[i] = o;
    }
}

// ---------------------------------------------------------------------------
// Kernel 2: fused GEMM (WMMA f16 -> f32) + row-wise argmax + coordinate gather.
// Block = 256 threads = 8 waves; each wave owns 32 rows (2 x 16-row tiles) of
// one batch and streams all 512 column tiles of desc_n.
// ---------------------------------------------------------------------------
__global__ void __launch_bounds__(256)
nn_match(const _Float16* __restrict__ ha,
         const _Float16* __restrict__ hb,
         const float* __restrict__ xy_n,
         const long long* __restrict__ n_r,
         const long long* __restrict__ n_n,
         float* __restrict__ out_coord)
{
    const int b    = blockIdx.y;
    const int tid  = threadIdx.x;
    const int wave = tid >> 5;
    const int lane = tid & 31;
    const int ln   = lane & 15;      // column / row-within-tile index
    const int hs   = lane >> 4;      // half-wave select (K split)
    const int R    = (blockIdx.x * 8 + wave) * 32;   // row base in this batch

    long long m64 = n_r[b] < n_n[b] ? n_r[b] : n_n[b];
    int mi = (int)(m64 < 0 ? 0 : (m64 > NN ? (long long)NN : m64));

    const _Float16* Ab = ha + (size_t)b * NN * DD;
    const _Float16* Bb = hb + (size_t)b * NN * DD;

    // A fragments: 2 row tiles x 2 K-blocks. A k order used here (same as B): lane pair (l,l+16)
    // shares row ln; halves [0..15] come from K = kb*32 + 16*hs + s
    // (contiguous runs -> plain b128 loads).
    v16h afrag[2][2];
#pragma unroll
    for (int rt = 0; rt < 2; ++rt) {
        const _Float16* ap = Ab + (size_t)(R + rt * 16 + ln) * DD;
#pragma unroll
        for (int kb = 0; kb < 2; ++kb) {
            v8h lo = *(const v8h*)(ap + kb * 32 + 16 * hs);
            v8h hi = *(const v8h*)(ap + kb * 32 + 16 * hs + 8);
            afrag[rt][kb] = __builtin_shufflevector(
                lo, hi, 0, 1, 2, 3, 4, 5, 6, 7, 8, 9, 10, 11, 12, 13, 14, 15);
        }
    }

    float bv[2][8];
    int   bi[2][8];
#pragma unroll
    for (int rt = 0; rt < 2; ++rt)
#pragma unroll
        for (int r = 0; r < 8; ++r) { bv[rt][r] = -__builtin_inff(); bi[rt][r] = 0; }

    const int ntf = mi >> 4;   // full (unmasked) column tiles

    for (int j = 0; j < ntf; ++j) {
        const int col = j * 16 + ln;
        // B fragment: column 'col' row of normalized desc_n; lane needs the
        // contiguous 16 halves at K = kb*32 + 16*hs  (32B aligned).
        const _Float16* bp = Bb + (size_t)col * DD + 16 * hs;
        __builtin_prefetch((const void*)(bp + 16 * DD), 0, 3);  // next tile
        v16h b0 = *(const v16h*)(bp);
        v16h b1 = *(const v16h*)(bp + 32);

#pragma unroll
        for (int rt = 0; rt < 2; ++rt) {
            v8f c = {};
            c = __builtin_amdgcn_wmma_f32_16x16x32_f16(false, afrag[rt][0], false, b0,
                                                       (short)0, c, false, false);
            c = __builtin_amdgcn_wmma_f32_16x16x32_f16(false, afrag[rt][1], false, b1,
                                                       (short)0, c, false, false);
#pragma unroll
            for (int r = 0; r < 8; ++r) {
                float v = c[r];
                if (v > bv[rt][r]) { bv[rt][r] = v; bi[rt][r] = col; }
            }
        }
    }

    if (mi & 15) {  // masked remainder tile (columns >= mi get -inf)
        const int col = ntf * 16 + ln;
        const _Float16* bp = Bb + (size_t)col * DD + 16 * hs;
        v16h b0 = *(const v16h*)(bp);
        v16h b1 = *(const v16h*)(bp + 32);
        const bool valid = col < mi;
#pragma unroll
        for (int rt = 0; rt < 2; ++rt) {
            v8f c = {};
            c = __builtin_amdgcn_wmma_f32_16x16x32_f16(false, afrag[rt][0], false, b0,
                                                       (short)0, c, false, false);
            c = __builtin_amdgcn_wmma_f32_16x16x32_f16(false, afrag[rt][1], false, b1,
                                                       (short)0, c, false, false);
#pragma unroll
            for (int r = 0; r < 8; ++r) {
                float v = valid ? c[r] : -__builtin_inff();
                if (v > bv[rt][r]) { bv[rt][r] = v; bi[rt][r] = col; }
            }
        }
    }

    // Cross-lane reduction over the 16 lanes of each half-wave.
    // Tie-break: smaller column index wins (argmax first-occurrence).
    const float2* xn = (const float2*)xy_n;
    float2* oc = (float2*)out_coord;
#pragma unroll
    for (int rt = 0; rt < 2; ++rt) {
#pragma unroll
        for (int r = 0; r < 8; ++r) {
            float v = bv[rt][r];
            int   i = bi[rt][r];
#pragma unroll
            for (int off = 1; off < 16; off <<= 1) {
                float ov = __shfl_xor(v, off, 32);
                int   oi = __shfl_xor(i, off, 32);
                if (ov > v || (ov == v && oi < i)) { v = ov; i = oi; }
            }
            if (ln == 0) {
                // C-layout: VGPR r = row r (lanes 0-15) / row r+8 (lanes 16-31)
                const int row = R + rt * 16 + hs * 8 + r;
                oc[(size_t)b * NN + row] = xn[(size_t)b * NN + i];
            }
        }
    }
}

// ---------------------------------------------------------------------------
extern "C" void kernel_launch(void* const* d_in, const int* in_sizes, int n_in,
                              void* d_out, int out_size, void* d_ws, size_t ws_size,
                              hipStream_t stream)
{
    const float*     xy_r   = (const float*)d_in[0];
    const float*     xy_n   = (const float*)d_in[1];
    const float*     desc_r = (const float*)d_in[2];
    const float*     desc_n = (const float*)d_in[3];
    const long long* n_r    = (const long long*)d_in[4];
    const long long* n_n    = (const long long*)d_in[5];
    float*           out    = (float*)d_out;

    _Float16* ha = (_Float16*)d_ws;                   // 4 MB
    _Float16* hb = ha + (size_t)BB * NN * DD;         // 4 MB

    // Phase 1: normalize + f16 convert + xy_r passthrough
    {
        int total = 2 * BB * NN;
        nn_pack<<<dim3((total + 255) / 256), 256, 0, stream>>>(
            xy_r, desc_r, desc_n, ha, hb, out);
    }
    // Phase 2: fused WMMA GEMM + argmax + gather
    {
        dim3 grid(NN / 256, BB);   // 8 waves * 32 rows = 256 rows per block
        nn_match<<<grid, 256, 0, stream>>>(
            ha, hb, xy_n, n_r, n_n, out + (size_t)BB * NN * 2);
    }
}
